// Linear3Bit_85452669321429
// MI455X (gfx1250) — matrix operand order repaired and checked by reading, not compiled, not run on hardware
//
#include <hip/hip_runtime.h>
#include <hip/hip_bf16.h>

// ---------------------------------------------------------------------------
// Fused 3-bit-dequant + bf16 WMMA GEMM for gfx1250 (MI455X)
//   out[T,O] = x[T,I] * W[O,I]^T + bias[O]
//   W: 3-bit codes, group=64 along I, per-group f32 scale
//   x tile staged via GLOBAL_LOAD_ASYNC_TO_LDS_B128 (ASYNCcnt path)
// ---------------------------------------------------------------------------

#define TOKENS 8192
#define IN_F   4096
#define OUT_F  4096
#define GROUPS_PER_ROW (IN_F / 64)   // 64
#define INTS_PER_GROUP 24            // 24 int32, each holding one byte

// Tiling
#define BM 128
#define BN 128
#define BK 64          // one quant group per K chunk
#define LDAF 68        // padded f32 row stride (272B) -> banks 4l..4l+3, conflict-free
#define LDB  72        // padded bf16 row stride (144B) -> conflict-free b128 reads

typedef __attribute__((ext_vector_type(16))) __bf16 v16bf;
typedef __attribute__((ext_vector_type(8)))  float  v8f;

union PackBF { __bf16 h[32]; uint4 q[4]; };
union FragBF { __bf16 h[16]; uint4 u[2]; v16bf v; };

__global__ __launch_bounds__(256) void
linear3bit_wmma_kernel(const float* __restrict__ x,
                       const int*   __restrict__ wq3,
                       const float* __restrict__ wscale,
                       const float* __restrict__ bias,
                       float*       __restrict__ out)
{
    __shared__ __align__(16) float  Asf[BM * LDAF];  // 34816 B, raw f32 via async DMA
    __shared__ __align__(16) __bf16 Bs [BN * LDB];   // 18432 B, dequantized bf16

    const int tid   = threadIdx.x;
    const int wave  = tid >> 5;        // 0..7 : M sub-tile owner
    const int lane  = tid & 31;
    const int lhalf = lane >> 4;       // 0/1
    const int lmod  = lane & 15;

    const int blockN = blockIdx.x;     // 0..31  (out-feature tile)
    const int blockM = blockIdx.y;     // 0..63  (token tile)

    // Loader mapping: 2 threads per row, each covers 32 elements of K
    const int rloc = tid >> 1;         // 0..127
    const int half = tid & 1;          // 0/1 -> k offset 0/32

    const int gRow = blockM * BM + rloc;            // token row for A
    const int gOut = blockN * BN + rloc;            // out-feature row for B

    // LDS byte address for this thread's async destination (low 32 bits of
    // the generic pointer hold the LDS offset on AMDGPU)
    const unsigned ldsA = (unsigned)(uintptr_t)(const void*)&Asf[rloc * LDAF + half * 32];

    v8f acc[8] = {};

    for (int kt = 0; kt < GROUPS_PER_ROW; ++kt) {
        // -------- stage A: async DMA x f32 tile straight into LDS --------
        {
            const unsigned long long gsrc =
                (unsigned long long)(uintptr_t)(x + (size_t)gRow * IN_F + kt * BK + half * 32);
#pragma unroll
            for (int i = 0; i < 8; ++i) {   // 8 x 16B per thread = 128B
                asm volatile("global_load_async_to_lds_b128 %0, %1, off"
                             :
                             : "v"(ldsA + i * 16), "v"(gsrc + (unsigned long long)(i * 16))
                             : "memory");
            }
        }
        // -------- stage B: dequant 3-bit -> bf16 into LDS ----------------
        {
            const size_t g = (size_t)gOut * GROUPS_PER_ROW + kt;
            const int4* wp = (const int4*)(wq3 + g * INTS_PER_GROUP + half * 12);
            int4 p0 = wp[0], p1 = wp[1], p2 = wp[2];
            int b[12];
            b[0] = p0.x & 255; b[1]  = p0.y & 255; b[2]  = p0.z & 255; b[3]  = p0.w & 255;
            b[4] = p1.x & 255; b[5]  = p1.y & 255; b[6]  = p1.z & 255; b[7]  = p1.w & 255;
            b[8] = p2.x & 255; b[9]  = p2.y & 255; b[10] = p2.z & 255; b[11] = p2.w & 255;
            const float sc = wscale[g];
            PackBF wb;
#pragma unroll
            for (int t = 0; t < 4; ++t) {
                const int b0 = b[3 * t], b1 = b[3 * t + 1], b2 = b[3 * t + 2];
                int c[8];
                c[0] = b0 & 7;
                c[1] = (b0 >> 3) & 7;
                c[2] = ((b0 >> 6) & 3) | ((b1 & 1) << 2);
                c[3] = (b1 >> 1) & 7;
                c[4] = (b1 >> 4) & 7;
                c[5] = ((b1 >> 7) & 1) | ((b2 & 3) << 1);
                c[6] = (b2 >> 2) & 7;
                c[7] = (b2 >> 5) & 7;
#pragma unroll
                for (int j = 0; j < 8; ++j) {
                    int q = c[j] - 3;
                    q = (q > 3) ? 3 : q;          // clip (only code 7 clips)
                    wb.h[8 * t + j] = (__bf16)((float)q * sc);
                }
            }
            uint4* dst = (uint4*)&Bs[rloc * LDB + half * 32];
#pragma unroll
            for (int i = 0; i < 4; ++i) dst[i] = wb.q[i];
        }

        // prefetch next K-chunk of packed weights
        if (kt + 1 < GROUPS_PER_ROW) {
            __builtin_prefetch(wq3 + ((size_t)gOut * GROUPS_PER_ROW + kt + 1) * INTS_PER_GROUP + half * 12, 0, 1);
        }

        // async DMA must land before anyone reads the A tile
        asm volatile("s_wait_asynccnt 0x0" ::: "memory");
        __syncthreads();

        // -------- compute: 2 K-steps x 8 N-tiles of WMMA -----------------
#pragma unroll
        for (int s = 0; s < 2; ++s) {
            // A fragment: 16x32 bf16 lane layout; source is f32 in LDS,
            // convert on the fly (v_cvt_pk_bf16_f32, co-executes with WMMA)
            const float4* ap =
                (const float4*)&Asf[(wave * 16 + lmod) * LDAF + s * 32 + lhalf * 8];
            float4 a0 = ap[0], a1 = ap[1];   // K +0..7  of this lane's half
            float4 a2 = ap[4], a3 = ap[5];   // K +16..23
            float fa[16] = { a0.x, a0.y, a0.z, a0.w,  a1.x, a1.y, a1.z, a1.w,
                             a2.x, a2.y, a2.z, a2.w,  a3.x, a3.y, a3.z, a3.w };
            FragBF af;
#pragma unroll
            for (int i = 0; i < 16; ++i) af.h[i] = (__bf16)fa[i];

#pragma unroll
            for (int nt = 0; nt < 8; ++nt) {
                FragBF bf;
                const uint4* bp =
                    (const uint4*)&Bs[(nt * 16 + lmod) * LDB + s * 32 + lhalf * 8];
                bf.u[0] = bp[0];
                bf.u[1] = bp[2];
                acc[nt] = __builtin_amdgcn_wmma_f32_16x16x32_bf16(
                    false, af.v, false, bf.v, (short)0, acc[nt], false, false);
            }
        }
        __syncthreads();
    }

    // -------- epilogue: bias + store -------------------------------------
    const int rowBase = blockM * BM + wave * 16 + lhalf * 8;   // M of acc[.][0]
    const int colBase = blockN * BN + lmod;
#pragma unroll
    for (int nt = 0; nt < 8; ++nt) {
        const int col = colBase + nt * 16;
        const float bv = bias[col];
#pragma unroll
        for (int r = 0; r < 8; ++r) {
            out[(size_t)(rowBase + r) * OUT_F + col] = acc[nt][r] + bv;
        }
    }
}

extern "C" void kernel_launch(void* const* d_in, const int* in_sizes, int n_in,
                              void* d_out, int out_size, void* d_ws, size_t ws_size,
                              hipStream_t stream) {
    (void)in_sizes; (void)n_in; (void)out_size; (void)d_ws; (void)ws_size;
    const float* x      = (const float*)d_in[0];
    const int*   wq3    = (const int*)d_in[1];
    const float* wscale = (const float*)d_in[2];
    const float* bias   = (const float*)d_in[3];
    float* out = (float*)d_out;

    dim3 grid(OUT_F / BN, TOKENS / BM);   // 32 x 64
    dim3 block(256);
    linear3bit_wmma_kernel<<<grid, block, 0, stream>>>(x, wq3, wscale, bias, out);
}
